// ReSkipBlockGroup_1537598292481
// MI455X (gfx1250) — matrix operand order repaired and checked by reading, not compiled, not run on hardware
//
#include <hip/hip_runtime.h>
#include <math.h>

// ---------------- problem constants (from reference) ----------------
#define L_N    2
#define D_N    2048
#define I_N    5632
#define H_N    16
#define B_N    2
#define T_N    1024
#define NBLK_N 4
#define BT_N   (B_N * T_N)      // 2048 tokens
#define DH_N   (D_N / H_N)      // 128 head dim
#define BH_N   (B_N * H_N)      // 32 batched heads
#define EPS_F  1e-6f

typedef __attribute__((ext_vector_type(16))) __bf16 v16bf;
typedef __attribute__((ext_vector_type(8)))  __bf16 v8bf;
typedef __attribute__((ext_vector_type(8)))  float  v8f;

// LDS byte offset of a generic pointer into shared memory (addrspace(3)
// pointer value IS the offset from the wave's LDS base).
__device__ __forceinline__ unsigned lds_off(const void* p) {
  return (unsigned)(unsigned long long)(__attribute__((address_space(3))) const char*)p;
}

// Async global->LDS 16B copy, GVS mode: per-lane LDS offset in VGPR,
// per-lane 32-bit global byte offset in VGPR, uniform 64-bit base in SGPRs.
// Tracked by ASYNCcnt.
__device__ __forceinline__ void async_ld16(unsigned ldsoff, unsigned goff,
                                           unsigned long long base) {
  asm volatile("global_load_async_to_lds_b128 %0, %1, %2 offset:0"
               :: "v"(ldsoff), "v"(goff), "s"(base)
               : "memory");
}

__device__ __forceinline__ void wait_async_le4() {
  asm volatile("s_wait_asynccnt 0x4" ::: "memory");
}
__device__ __forceinline__ void wait_async_0() {
  asm volatile("s_wait_asynccnt 0x0" ::: "memory");
}

// ---------------- block reductions (blockDim.x == 256) ----------------
__device__ __forceinline__ float blk_sum(float v, volatile float* s) {
  int t = threadIdx.x;
  s[t] = v; __syncthreads();
  if (t < 128) s[t] += s[t + 128]; __syncthreads();
  if (t < 64)  s[t] += s[t + 64];  __syncthreads();
  if (t < 32)  s[t] += s[t + 32];  __syncthreads();
  if (t < 16)  s[t] += s[t + 16];  __syncthreads();
  if (t < 8)   s[t] += s[t + 8];   __syncthreads();
  if (t < 4)   s[t] += s[t + 4];   __syncthreads();
  if (t < 2)   s[t] += s[t + 2];   __syncthreads();
  if (t < 1)   s[t] += s[t + 1];   __syncthreads();
  float r = s[0]; __syncthreads();
  return r;
}

__device__ __forceinline__ float blk_max(float v, volatile float* s) {
  int t = threadIdx.x;
  s[t] = v; __syncthreads();
  if (t < 128) s[t] = fmaxf(s[t], s[t + 128]); __syncthreads();
  if (t < 64)  s[t] = fmaxf(s[t], s[t + 64]);  __syncthreads();
  if (t < 32)  s[t] = fmaxf(s[t], s[t + 32]);  __syncthreads();
  if (t < 16)  s[t] = fmaxf(s[t], s[t + 16]);  __syncthreads();
  if (t < 8)   s[t] = fmaxf(s[t], s[t + 8]);   __syncthreads();
  if (t < 4)   s[t] = fmaxf(s[t], s[t + 4]);   __syncthreads();
  if (t < 2)   s[t] = fmaxf(s[t], s[t + 2]);   __syncthreads();
  if (t < 1)   s[t] = fmaxf(s[t], s[t + 1]);   __syncthreads();
  float r = s[0]; __syncthreads();
  return r;
}

// ---------------- batched bf16 WMMA GEMM (nt only) ----------------
// C[M,N] (f32) = A[M,K] (bf16 row-major) x B^T, B stored [N,K] row-major.
// Block tile 128x128, BK=32, 256 threads = 8 waves (wave32); each wave owns
// a 64x32 sub-tile (4 M-frags x 2 N-frags -> 8 WMMAs per K-step).
// Staging: double-buffered LDS filled with global_load_async_to_lds_b128,
// pipelined one K-stage ahead with s_wait_asynccnt.
// Requires M%128==0, N%128==0, K%32==0; rows 16B-aligned (K%8==0).
__global__ void __launch_bounds__(256)
gemm_bf16_wmma_nt(const __bf16* __restrict__ A, const __bf16* __restrict__ Bm,
                  float* __restrict__ C, int M, int N, int K,
                  long sA, long sB, long sC) {
  (void)M;
  const int bz = blockIdx.z;
  A  += (long)bz * sA;
  Bm += (long)bz * sB;
  C  += (long)bz * sC;

  const int m0   = blockIdx.y * 128;
  const int n0   = blockIdx.x * 128;
  const int tid  = threadIdx.x;
  const int lane = tid & 31;
  const int wid  = tid >> 5;
  const int mw   = (wid >> 2) * 64;  // wave M base: 0 or 64
  const int nw   = (wid & 3) * 32;   // wave N base: 0..96
  const int lr   = lane & 15;        // row/col within fragment
  const int lh   = lane >> 4;        // lane half (K-group select)

  __shared__ __align__(16) __bf16 As[2][128][40];  // [buf][M][K+pad]
  __shared__ __align__(16) __bf16 Bs[2][128][40];  // [buf][N][K+pad]

  // fixed per-thread staging chunk coords: 512 chunks of 8 bf16 per tile
  const int r0 = tid >> 2,         kc0 = (tid & 3) * 8;
  const int r1 = (tid + 256) >> 2, kc1 = ((tid + 256) & 3) * 8;

  const unsigned long long abase = (unsigned long long)A;
  const unsigned long long bbase = (unsigned long long)Bm;

  auto issue_stage = [&](int k0, int buf) {
    async_ld16(lds_off(&As[buf][r0][kc0]),
               (unsigned)((((long)(m0 + r0)) * K + k0 + kc0) * 2), abase);
    async_ld16(lds_off(&As[buf][r1][kc1]),
               (unsigned)((((long)(m0 + r1)) * K + k0 + kc1) * 2), abase);
    async_ld16(lds_off(&Bs[buf][r0][kc0]),
               (unsigned)((((long)(n0 + r0)) * K + k0 + kc0) * 2), bbase);
    async_ld16(lds_off(&Bs[buf][r1][kc1]),
               (unsigned)((((long)(n0 + r1)) * K + k0 + kc1) * 2), bbase);
  };

  v8f acc[4][2];
#pragma unroll
  for (int i = 0; i < 4; ++i)
#pragma unroll
    for (int j = 0; j < 2; ++j)
#pragma unroll
      for (int r = 0; r < 8; ++r) acc[i][j][r] = 0.0f;

  const int S = K / 32;
  issue_stage(0, 0);

  for (int s = 0; s < S; ++s) {
    if (s + 1 < S) {
      issue_stage((s + 1) * 32, (s + 1) & 1);
      wait_async_le4();  // own wave's stage-s group done (in-order completion)
    } else {
      wait_async_0();
    }
    __syncthreads();  // all waves' stage-s data visible

    const int b = s & 1;
    v16bf fa[4], fb[2];
#pragma unroll
    for (int mf = 0; mf < 4; ++mf) {
      int m   = mw + mf * 16 + lr;
      v8bf lo = *(const v8bf*)&As[b][m][lh * 8];       // K = lh*8 + 0..7
      v8bf hi = *(const v8bf*)&As[b][m][16 + lh * 8];  // K = 16 + lh*8 + 0..7
      fa[mf]  = __builtin_shufflevector(lo, hi, 0, 1, 2, 3, 4, 5, 6, 7, 8, 9,
                                        10, 11, 12, 13, 14, 15);
    }
#pragma unroll
    for (int nf = 0; nf < 2; ++nf) {
      int n   = nw + nf * 16 + lr;
      v8bf lo = *(const v8bf*)&Bs[b][n][lh * 8];       // K = lh*8 + 0..7
      v8bf hi = *(const v8bf*)&Bs[b][n][16 + lh * 8];  // K = 16 + lh*8 + 0..7
      fb[nf]  = __builtin_shufflevector(lo, hi, 0, 1, 2, 3, 4, 5, 6, 7, 8, 9,
                                        10, 11, 12, 13, 14, 15);
    }
#pragma unroll
    for (int mf = 0; mf < 4; ++mf)
#pragma unroll
      for (int nf = 0; nf < 2; ++nf)
        acc[mf][nf] = __builtin_amdgcn_wmma_f32_16x16x32_bf16(
            false, fa[mf], false, fb[nf], (short)0, acc[mf][nf], false, false);

    __syncthreads();  // done reading buf b before it is overwritten at s+2
  }

  // ---- store C (VGPR r: M = lh*8 + r, N = lr) ----
#pragma unroll
  for (int mf = 0; mf < 4; ++mf)
#pragma unroll
    for (int nf = 0; nf < 2; ++nf)
#pragma unroll
      for (int r = 0; r < 8; ++r) {
        int m = m0 + mw + mf * 16 + lh * 8 + r;
        int n = n0 + nw + nf * 16 + lr;
        C[(long)m * N + n] = acc[mf][nf][r];
      }
}

// ---------------- elementwise / reduction kernels ----------------
// f32 [L,K,N] -> bf16 [L,N,K]  (weight convert + transpose, once per launch)
__global__ void cvt_t_bf16(const float* __restrict__ in,
                           __bf16* __restrict__ out, int Kd, int Nd, int Lc) {
  long total  = (long)Lc * Kd * Nd;
  long stride = (long)gridDim.x * 256;
  for (long e = (long)blockIdx.x * 256 + threadIdx.x; e < total; e += stride) {
    int n  = (int)(e % Nd);
    long r = e / Nd;
    int k  = (int)(r % Kd);
    int l  = (int)(r / Kd);
    out[((long)l * Nd + n) * (long)Kd + k] = (__bf16)in[e];
  }
}

__global__ void qeff_kernel(const float* wqa, const float* kna,
                            const float* wqm, const float* knm, float* qeff) {
  int d = blockIdx.x * 256 + threadIdx.x;
  if (d >= D_N) return;
#pragma unroll
  for (int i = 0; i < L_N; ++i) {
    qeff[(2 * i + 0) * D_N + d] = wqa[i * D_N + d] * kna[i * D_N + d];
    qeff[(2 * i + 1) * D_N + d] = wqm[i * D_N + d] * knm[i * D_N + d];
  }
}

// scores[q][n][bt] = (q_eff[q] . rms(states[n,bt])) / sqrt(D)
__global__ void phase1_scores(const float* __restrict__ states,
                              const float* __restrict__ qeff,
                              float* __restrict__ scores) {
  __shared__ float red[256];
  int n  = blockIdx.x / BT_N;
  int bt = blockIdx.x % BT_N;
  const float* x = states + ((long)n * BT_N + bt) * (long)D_N;
  float ss = 0.f, d0 = 0.f, d1 = 0.f, d2 = 0.f, d3 = 0.f;
  for (int d = threadIdx.x; d < D_N; d += 256) {
    float v = x[d];
    ss += v * v;
    d0 += qeff[d] * v;
    d1 += qeff[D_N + d] * v;
    d2 += qeff[2 * D_N + d] * v;
    d3 += qeff[3 * D_N + d] * v;
  }
  ss = blk_sum(ss, red);
  d0 = blk_sum(d0, red);
  d1 = blk_sum(d1, red);
  d2 = blk_sum(d2, red);
  d3 = blk_sum(d3, red);
  if (threadIdx.x == 0) {
    float sc = rsqrtf(ss / D_N + EPS_F) / sqrtf((float)D_N);
    scores[((long)0 * NBLK_N + n) * BT_N + bt] = d0 * sc;
    scores[((long)1 * NBLK_N + n) * BT_N + bt] = d1 * sc;
    scores[((long)2 * NBLK_N + n) * BT_N + bt] = d2 * sc;
    scores[((long)3 * NBLK_N + n) * BT_N + bt] = d3 * sc;
  }
}

__global__ void phase1_stats(const float* __restrict__ scores, float* mx,
                             float* lse, float* ex) {
  int i = blockIdx.x * 256 + threadIdx.x;
  if (i >= 4 * BT_N) return;
  int q = i / BT_N, bt = i % BT_N;
  float m = -3.4e38f;
  for (int n = 0; n < NBLK_N; ++n)
    m = fmaxf(m, scores[((long)q * NBLK_N + n) * BT_N + bt]);
  float s = 0.f;
  for (int n = 0; n < NBLK_N; ++n) {
    float e = expf(scores[((long)q * NBLK_N + n) * BT_N + bt] - m);
    ex[((long)q * NBLK_N + n) * BT_N + bt] = e;
    s += e;
  }
  mx[(long)q * BT_N + bt]  = m;
  lse[(long)q * BT_N + bt] = s;
}

// outs[q][bt][d] = sum_n ex[q][n][bt] * states[n][bt][d]   (unnormalized)
__global__ void phase1_outs(const float* __restrict__ states,
                            const float* __restrict__ ex,
                            float* __restrict__ outs) {
  int bt = blockIdx.x;
  float w[4][NBLK_N];
#pragma unroll
  for (int q = 0; q < 4; ++q)
#pragma unroll
    for (int n = 0; n < NBLK_N; ++n)
      w[q][n] = ex[((long)q * NBLK_N + n) * BT_N + bt];
  for (int d = threadIdx.x; d < D_N; d += 256) {
    float a0 = 0.f, a1 = 0.f, a2 = 0.f, a3 = 0.f;
#pragma unroll
    for (int n = 0; n < NBLK_N; ++n) {
      float v = states[((long)n * BT_N + bt) * D_N + d];
      a0 += w[0][n] * v;
      a1 += w[1][n] * v;
      a2 += w[2][n] * v;
      a3 += w[3][n] * v;
    }
    outs[((long)0 * BT_N + bt) * D_N + d] = a0;
    outs[((long)1 * BT_N + bt) * D_N + d] = a1;
    outs[((long)2 * BT_N + bt) * D_N + d] = a2;
    outs[((long)3 * BT_N + bt) * D_N + d] = a3;
  }
}

// merge_with_partial_block (exactly mirrors reference arithmetic)
__global__ void merge_kernel(int hasPartial, const float* __restrict__ outsq,
                             const float* __restrict__ mxq,
                             const float* __restrict__ lseq,
                             const float* __restrict__ qrow,
                             const float* __restrict__ partial,
                             float* __restrict__ dst) {
  __shared__ float red[256];
  int bt            = blockIdx.x;
  const float* orow = outsq + (long)bt * D_N;
  float lse         = lseq[bt];
  if (!hasPartial) {
    float inv = 1.f / lse;
    for (int d = threadIdx.x; d < D_N; d += 256)
      dst[(long)bt * D_N + d] = orow[d] * inv;
    return;
  }
  const float* prow = partial + (long)bt * D_N;
  float ss = 0.f, dot = 0.f;
  for (int d = threadIdx.x; d < D_N; d += 256) {
    float v = prow[d];
    ss += v * v;
    dot += qrow[d] * v;
  }
  ss  = blk_sum(ss, red);
  dot = blk_sum(dot, red);
  float rstd = rsqrtf(ss / D_N + EPS_F);
  float ps   = dot * rstd / sqrtf((float)D_N);
  float mx   = mxq[bt];
  float m2   = fmaxf(mx, ps);
  float c1   = expf(mx - m2), c2 = expf(ps - m2);
  float den  = c1 * lse + c2;
  float w1   = c1 * lse / den;
  float w2   = c2 / den;
  for (int d = threadIdx.x; d < D_N; d += 256)
    dst[(long)bt * D_N + d] = orow[d] * w1 + prow[d] * w2;
}

__global__ void rmsnorm_bf16(const float* __restrict__ x,
                             const float* __restrict__ w,
                             __bf16* __restrict__ y) {
  __shared__ float red[256];
  int r           = blockIdx.x;
  const float* xr = x + (long)r * D_N;
  float ss        = 0.f;
  for (int d = threadIdx.x; d < D_N; d += 256) {
    float v = xr[d];
    ss += v * v;
  }
  ss         = blk_sum(ss, red);
  float rstd = rsqrtf(ss / D_N + EPS_F);
  for (int d = threadIdx.x; d < D_N; d += 256)
    y[(long)r * D_N + d] = (__bf16)(xr[d] * rstd * w[d]);
}

__global__ void rope_kernel(float* __restrict__ x) {  // x: [B,T,H,dh], in-place
  int i           = blockIdx.x * 256 + threadIdx.x;
  const int total = B_N * T_N * H_N * (DH_N / 2);
  if (i >= total) return;
  int j = i % (DH_N / 2);
  int r = i / (DH_N / 2);
  int h = r % H_N;
  r /= H_N;
  int t     = r % T_N;
  int b     = r / T_N;
  float inv = expf(-(2.f * j / (float)DH_N) * logf(10000.f));
  float fr  = (float)t * inv;
  float c = cosf(fr), s = sinf(fr);
  long base = (((long)b * T_N + t) * H_N + h) * DH_N;
  float x1 = x[base + j], x2 = x[base + DH_N / 2 + j];
  x[base + j]            = x1 * c - x2 * s;
  x[base + DH_N / 2 + j] = x2 * c + x1 * s;
}

// [B,T,H,dh] f32 -> [B*H, T, dh] bf16   (for Q and K)
__global__ void trans_bhtd_bf16(const float* __restrict__ in,
                                __bf16* __restrict__ out) {
  long e = (long)blockIdx.x * 256 + threadIdx.x;
  if (e >= (long)BT_N * D_N) return;
  int d  = (int)(e % DH_N);
  long r = e / DH_N;
  int h  = (int)(r % H_N);
  r /= H_N;
  int t = (int)(r % T_N);
  int b = (int)(r / T_N);
  out[(((long)(b * H_N + h) * T_N) + t) * DH_N + d] = (__bf16)in[e];
}

// [B,T,H,dh] f32 -> [B*H, dh, T] bf16   (V transposed, for nt AV GEMM)
__global__ void trans_bhdt_bf16(const float* __restrict__ in,
                                __bf16* __restrict__ out) {
  long e = (long)blockIdx.x * 256 + threadIdx.x;
  if (e >= (long)BT_N * D_N) return;
  int d  = (int)(e % DH_N);
  long r = e / DH_N;
  int h  = (int)(r % H_N);
  r /= H_N;
  int t = (int)(r % T_N);
  int b = (int)(r / T_N);
  out[(((long)(b * H_N + h) * DH_N) + d) * T_N + t] = (__bf16)in[e];
}

// [B,H,T,dh] f32 -> [B,T,H*dh] bf16
__global__ void trans_btHd_bf16(const float* __restrict__ in,
                                __bf16* __restrict__ out) {
  long e = (long)blockIdx.x * 256 + threadIdx.x;
  if (e >= (long)BT_N * D_N) return;
  int d  = (int)(e % DH_N);
  long r = e / DH_N;
  int t  = (int)(r % T_N);
  r /= T_N;
  int h = (int)(r % H_N);
  int b = (int)(r / H_N);
  out[(((long)b * T_N + t) * H_N + h) * DH_N + d] = (__bf16)in[e];
}

// causal softmax over rows of S [BH*T, T], scaled by 1/sqrt(dh), bf16 out
__global__ void softmax_causal(const float* __restrict__ S,
                               __bf16* __restrict__ P) {
  __shared__ float red[256];
  long row          = blockIdx.x;
  int qpos          = (int)(row % T_N);
  const float* srow = S + row * T_N;
  __bf16* prow      = P + row * T_N;
  const float scale = rsqrtf((float)DH_N);
  float vals[T_N / 256];
  float vmax = -3.4e38f;
#pragma unroll
  for (int i = 0; i < T_N / 256; ++i) {
    int c   = threadIdx.x + i * 256;
    float v = (c <= qpos) ? srow[c] * scale : -1e30f;
    vals[i] = v;
    vmax    = fmaxf(vmax, v);
  }
  vmax      = blk_max(vmax, red);
  float sum = 0.f;
#pragma unroll
  for (int i = 0; i < T_N / 256; ++i) {
    float e = expf(vals[i] - vmax);
    vals[i] = e;
    sum += e;
  }
  sum       = blk_sum(sum, red);
  float inv = 1.f / sum;
#pragma unroll
  for (int i = 0; i < T_N / 256; ++i) {
    int c   = threadIdx.x + i * 256;
    prow[c] = (__bf16)(vals[i] * inv);
  }
}

__global__ void silu_mul_bf16(const float* __restrict__ g,
                              const float* __restrict__ u,
                              __bf16* __restrict__ out) {
  long e = (long)blockIdx.x * 256 + threadIdx.x;
  if (e >= (long)BT_N * I_N) return;
  float gv = g[e];
  float s  = gv / (1.f + expf(-gv));
  out[e]   = (__bf16)(s * u[e]);
}

// partial = mask*attn + (1-mask)*block_states[*idx]   (first blend)
__global__ void blend0_kernel(const float* __restrict__ attn,
                              const float* __restrict__ states,
                              const int* __restrict__ idxp,
                              const unsigned char* __restrict__ mask,
                              float* __restrict__ partial) {
  long e = (long)blockIdx.x * 256 + threadIdx.x;
  if (e >= (long)BT_N * D_N) return;
  int b      = (int)(e / ((long)T_N * D_N));
  float m    = mask[b] ? 1.f : 0.f;
  long base  = (long)(*idxp) * (long)BT_N * D_N;
  partial[e] = m * attn[e] + (1.f - m) * states[base + e];
}

// dst = in + mask*delta
__global__ void addmask_kernel(const float* __restrict__ in,
                               const float* __restrict__ delta,
                               const unsigned char* __restrict__ mask,
                               float* __restrict__ dst) {
  long e = (long)blockIdx.x * 256 + threadIdx.x;
  if (e >= (long)BT_N * D_N) return;
  int b   = (int)(e / ((long)T_N * D_N));
  float m = mask[b] ? 1.f : 0.f;
  dst[e]  = in[e] + m * delta[e];
}

// ---------------- launch orchestration ----------------
extern "C" void kernel_launch(void* const* d_in, const int* in_sizes, int n_in,
                              void* d_out, int out_size, void* d_ws,
                              size_t ws_size, hipStream_t stream) {
  (void)in_sizes; (void)n_in; (void)out_size; (void)ws_size;

  const float* bs            = (const float*)d_in[0];
  const unsigned char* amask = (const unsigned char*)d_in[1];
  const float* wqa           = (const float*)d_in[2];
  const float* kna           = (const float*)d_in[3];
  const float* wqm           = (const float*)d_in[4];
  const float* knm           = (const float*)d_in[5];
  const float* attn_nw       = (const float*)d_in[6];
  const float* Wq            = (const float*)d_in[7];
  const float* Wk            = (const float*)d_in[8];
  const float* Wv            = (const float*)d_in[9];
  const float* Wo            = (const float*)d_in[10];
  const float* mlp_nw        = (const float*)d_in[11];
  const float* Wg            = (const float*)d_in[12];
  const float* Wu            = (const float*)d_in[13];
  const float* Wd            = (const float*)d_in[14];
  const int* idxp            = (const int*)d_in[15];
  float* out                 = (float*)d_out;

  const long DD  = (long)D_N * D_N;
  const long DI  = (long)D_N * I_N;
  const long BTD = (long)BT_N * D_N;
  const long BTI = (long)BT_N * I_N;

  char* ws   = (char*)d_ws;
  size_t off = 0;
  auto alloc = [&](size_t bytes) -> char* {
    char* p = ws + off;
    off     = (off + bytes + 255) & ~(size_t)255;
    return p;
  };

  float* qeff    = (float*)alloc(4 * D_N * 4);
  float* scores  = (float*)alloc((size_t)4 * NBLK_N * BT_N * 4);
  float* mx      = (float*)alloc((size_t)4 * BT_N * 4);
  float* lse     = (float*)alloc((size_t)4 * BT_N * 4);
  float* ex      = (float*)alloc((size_t)4 * NBLK_N * BT_N * 4);
  float* outs    = (float*)alloc((size_t)4 * BTD * 4);
  float* partial = (float*)alloc((size_t)BTD * 4);
  float* merged  = (float*)alloc((size_t)BTD * 4);
  __bf16* xnb    = (__bf16*)alloc((size_t)BTD * 2);
  float* qf      = (float*)alloc((size_t)BTD * 4);
  float* kf      = (float*)alloc((size_t)BTD * 4);
  float* vf      = (float*)alloc((size_t)BTD * 4);
  __bf16* qhb    = (__bf16*)alloc((size_t)BTD * 2);
  __bf16* khb    = (__bf16*)alloc((size_t)BTD * 2);
  __bf16* vhb    = (__bf16*)alloc((size_t)BTD * 2);  // [BH, dh, T]
  float* S       = (float*)alloc((size_t)BH_N * T_N * T_N * 4);
  __bf16* Pb     = (__bf16*)alloc((size_t)BH_N * T_N * T_N * 2);
  float* Of      = (float*)alloc((size_t)BTD * 4);
  __bf16* aob    = (__bf16*)alloc((size_t)BTD * 2);
  float* tmpOut  = (float*)alloc((size_t)BTD * 4);
  float* gf      = (float*)alloc((size_t)BTI * 4);
  float* uf      = (float*)alloc((size_t)BTI * 4);
  __bf16* smb    = (__bf16*)alloc((size_t)BTI * 2);
  __bf16* wqb    = (__bf16*)alloc((size_t)L_N * DD * 2);  // [L,N,K] transposed
  __bf16* wkb    = (__bf16*)alloc((size_t)L_N * DD * 2);
  __bf16* wvb    = (__bf16*)alloc((size_t)L_N * DD * 2);
  __bf16* wob    = (__bf16*)alloc((size_t)L_N * DD * 2);
  __bf16* wgb    = (__bf16*)alloc((size_t)L_N * DI * 2);
  __bf16* wub    = (__bf16*)alloc((size_t)L_N * DI * 2);
  __bf16* wdb    = (__bf16*)alloc((size_t)L_N * DI * 2);

  // ---- weight down-conversion to bf16 + transpose to [N,K] ----
  cvt_t_bf16<<<2048, 256, 0, stream>>>(Wq, wqb, D_N, D_N, L_N);
  cvt_t_bf16<<<2048, 256, 0, stream>>>(Wk, wkb, D_N, D_N, L_N);
  cvt_t_bf16<<<2048, 256, 0, stream>>>(Wv, wvb, D_N, D_N, L_N);
  cvt_t_bf16<<<2048, 256, 0, stream>>>(Wo, wob, D_N, D_N, L_N);
  cvt_t_bf16<<<2048, 256, 0, stream>>>(Wg, wgb, D_N, I_N, L_N);
  cvt_t_bf16<<<2048, 256, 0, stream>>>(Wu, wub, D_N, I_N, L_N);
  cvt_t_bf16<<<2048, 256, 0, stream>>>(Wd, wdb, I_N, D_N, L_N);

  // ---- phase 1: router over completed blocks ----
  qeff_kernel<<<(D_N + 255) / 256, 256, 0, stream>>>(wqa, kna, wqm, knm, qeff);
  phase1_scores<<<NBLK_N * BT_N, 256, 0, stream>>>(bs, qeff, scores);
  phase1_stats<<<(4 * BT_N + 255) / 256, 256, 0, stream>>>(scores, mx, lse, ex);
  phase1_outs<<<BT_N, 256, 0, stream>>>(bs, ex, outs);

  const int nElemBlk  = (int)((BTD + 255) / 256);
  const int nElemBlkI = (int)((BTI + 255) / 256);
  const dim3 gD(16, 16, 1);                  // 2048x2048, K variable
  const dim3 gGate(I_N / 128, 16, 1);        // 2048x5632
  const dim3 gQK(T_N / 128, T_N / 128, BH_N);
  const dim3 gAV(DH_N / 128, T_N / 128, BH_N);

  for (int i = 0; i < L_N; ++i) {
    // ===== attention sublayer =====
    merge_kernel<<<BT_N, 256, 0, stream>>>(
        i == 0 ? 0 : 1, outs + (long)(2 * i) * BTD, mx + (long)(2 * i) * BT_N,
        lse + (long)(2 * i) * BT_N, qeff + (long)(2 * i) * D_N, partial, merged);
    rmsnorm_bf16<<<BT_N, 256, 0, stream>>>(merged, attn_nw + (long)i * D_N, xnb);

    gemm_bf16_wmma_nt<<<gD, 256, 0, stream>>>(xnb, wqb + (long)i * DD, qf,
                                              BT_N, D_N, D_N, 0, 0, 0);
    gemm_bf16_wmma_nt<<<gD, 256, 0, stream>>>(xnb, wkb + (long)i * DD, kf,
                                              BT_N, D_N, D_N, 0, 0, 0);
    gemm_bf16_wmma_nt<<<gD, 256, 0, stream>>>(xnb, wvb + (long)i * DD, vf,
                                              BT_N, D_N, D_N, 0, 0, 0);

    const int nRope = (B_N * T_N * H_N * (DH_N / 2) + 255) / 256;
    rope_kernel<<<nRope, 256, 0, stream>>>(qf);
    rope_kernel<<<nRope, 256, 0, stream>>>(kf);

    trans_bhtd_bf16<<<nElemBlk, 256, 0, stream>>>(qf, qhb);
    trans_bhtd_bf16<<<nElemBlk, 256, 0, stream>>>(kf, khb);
    trans_bhdt_bf16<<<nElemBlk, 256, 0, stream>>>(vf, vhb);

    // S = Q x K^T  (per batched head), causal softmax -> P (bf16)
    gemm_bf16_wmma_nt<<<gQK, 256, 0, stream>>>(
        qhb, khb, S, T_N, T_N, DH_N, (long)T_N * DH_N, (long)T_N * DH_N,
        (long)T_N * T_N);
    softmax_causal<<<BH_N * T_N, 256, 0, stream>>>(S, Pb);
    // O = P x V   (V held transposed [dh, T] -> nt GEMM)
    gemm_bf16_wmma_nt<<<gAV, 256, 0, stream>>>(
        Pb, vhb, Of, T_N, DH_N, T_N, (long)T_N * T_N, (long)DH_N * T_N,
        (long)T_N * DH_N);
    trans_btHd_bf16<<<nElemBlk, 256, 0, stream>>>(Of, aob);
    // attn_out = O x Wo^T(stored)
    gemm_bf16_wmma_nt<<<gD, 256, 0, stream>>>(aob, wob + (long)i * DD, tmpOut,
                                              BT_N, D_N, D_N, 0, 0, 0);

    if (i == 0)
      blend0_kernel<<<nElemBlk, 256, 0, stream>>>(tmpOut, bs, idxp, amask,
                                                  partial);
    else
      addmask_kernel<<<nElemBlk, 256, 0, stream>>>(partial, tmpOut, amask,
                                                   partial);

    // ===== MLP sublayer =====
    merge_kernel<<<BT_N, 256, 0, stream>>>(
        1, outs + (long)(2 * i + 1) * BTD, mx + (long)(2 * i + 1) * BT_N,
        lse + (long)(2 * i + 1) * BT_N, qeff + (long)(2 * i + 1) * D_N, partial,
        merged);
    rmsnorm_bf16<<<BT_N, 256, 0, stream>>>(merged, mlp_nw + (long)i * D_N, xnb);

    gemm_bf16_wmma_nt<<<gGate, 256, 0, stream>>>(xnb, wgb + (long)i * DI, gf,
                                                 BT_N, I_N, D_N, 0, 0, 0);
    gemm_bf16_wmma_nt<<<gGate, 256, 0, stream>>>(xnb, wub + (long)i * DI, uf,
                                                 BT_N, I_N, D_N, 0, 0, 0);
    silu_mul_bf16<<<nElemBlkI, 256, 0, stream>>>(gf, uf, smb);
    gemm_bf16_wmma_nt<<<gD, 256, 0, stream>>>(smb, wdb + (long)i * DI, tmpOut,
                                              BT_N, D_N, I_N, 0, 0, 0);

    float* dst = (i == L_N - 1) ? out : partial;
    addmask_kernel<<<nElemBlk, 256, 0, stream>>>(partial, tmpOut, amask, dst);
  }
}
